// G2ANet_65128884077157
// MI455X (gfx1250) — matrix operand order repaired and checked by reading, not compiled, not run on hardware
//
#include <hip/hip_runtime.h>
#include <hip/hip_bf16.h>

typedef __attribute__((ext_vector_type(16))) _Float16 v16h;
typedef __attribute__((ext_vector_type(8)))  _Float16 v8h;
typedef __attribute__((ext_vector_type(8)))  float    v8f;

#define SIZE   8192        // B * N_AGENTS
#define BSZ    1024
#define NAG    8
#define LL     7
#define HH     256
#define AA     128
#define INDIM  128
#define NACT   32
#define H3     768
#define H2     512
#define TAU_F  0.01f

// ---------------------------------------------------------------------------
// WMMA fragment loaders (per ISA 16-bit layouts)
//   A[M,K] row-major: lane (row=lr) holds K chunks at k0+g*8 and k0+16+g*8
//   B = W^T with W[N,K] row-major: lane (col=lr) holds K chunks at k0+g*8 and k0+16+g*8
// ---------------------------------------------------------------------------
__device__ __forceinline__ v16h frag_a(const _Float16* row, int k0, int g) {
  v8h lo = *(const v8h*)(row + k0 + g * 8);
  v8h hi = *(const v8h*)(row + k0 + 16 + g * 8);
  return __builtin_shufflevector(lo, hi, 0,1,2,3,4,5,6,7,8,9,10,11,12,13,14,15);
}
__device__ __forceinline__ v16h frag_b(const _Float16* row, int k0, int g) {
  v8h lo = *(const v8h*)(row + k0 + g * 8);
  v8h hi = *(const v8h*)(row + k0 + 16 + g * 8);
  return __builtin_shufflevector(lo, hi, 0,1,2,3,4,5,6,7,8,9,10,11,12,13,14,15);
}

// ---------------------------------------------------------------------------
// WMMA GEMM: C[M,N] = act( A[M,K](f16) @ W[N,K]^T(f16) + bias )
// Wave tile: 32 rows x NT*16 cols (2 M-tiles, NT N-tiles -> 2*NT wmma / K-slice)
// Block = 128 threads (4 waves, consecutive 32-row strips); grid.x=M/128,
// grid.y=N/(NT*16). All dims divide exactly; EXEC stays all-1s (no divergence).
// ---------------------------------------------------------------------------
template<int NT>
__global__ __launch_bounds__(128)
void wmma_gemm(const _Float16* __restrict__ A, const _Float16* __restrict__ Bw,
               const float* __restrict__ bias, float* __restrict__ Cf,
               _Float16* __restrict__ Ch, int K, int ldc, int ldch, int act) {
  const int lane = threadIdx.x & 31;
  const int wave = threadIdx.x >> 5;
  const int m0   = blockIdx.x * 128 + wave * 32;
  const int n0   = blockIdx.y * (NT * 16);
  const int g    = lane >> 4;
  const int lr   = lane & 15;

  const _Float16* arow0 = A + (size_t)(m0 + lr) * K;
  const _Float16* arow1 = A + (size_t)(m0 + 16 + lr) * K;
  const _Float16* brow[NT];
#pragma unroll
  for (int t = 0; t < NT; ++t) brow[t] = Bw + (size_t)(n0 + 16 * t + lr) * K;

  const v8f zero = {0,0,0,0,0,0,0,0};
  v8f acc[2][NT];
#pragma unroll
  for (int i = 0; i < 2; ++i)
#pragma unroll
    for (int t = 0; t < NT; ++t) acc[i][t] = zero;

  for (int k0 = 0; k0 < K; k0 += 32) {
    v16h a0 = frag_a(arow0, k0, g);
    v16h a1 = frag_a(arow1, k0, g);
#pragma unroll
    for (int t = 0; t < NT; ++t) {
      v16h b = frag_b(brow[t], k0, g);
      acc[0][t] = __builtin_amdgcn_wmma_f32_16x16x32_f16(false, a0, false, b, (short)0, acc[0][t], false, false);
      acc[1][t] = __builtin_amdgcn_wmma_f32_16x16x32_f16(false, a1, false, b, (short)0, acc[1][t], false, false);
    }
  }

#pragma unroll
  for (int t = 0; t < NT; ++t) {
    const int n = n0 + 16 * t + lr;
    const float bb = bias ? bias[n] : 0.f;
#pragma unroll
    for (int i = 0; i < 2; ++i) {
#pragma unroll
      for (int r = 0; r < 8; ++r) {
        const int m = m0 + 16 * i + r + 8 * g;
        float v = acc[i][t][r] + bb;
        if (act == 1) v = fmaxf(v, 0.f);
        if (Cf) Cf[(size_t)m * ldc + n] = v;
        if (Ch) Ch[(size_t)m * ldch + n] = (_Float16)v;
      }
    }
  }
}

// ---------------------------------------------------------------------------
// Elementwise helpers
// ---------------------------------------------------------------------------
__global__ void cvt_f16(const float* __restrict__ s, _Float16* __restrict__ d, int n) {
  int i = blockIdx.x * blockDim.x + threadIdx.x;
  if (i < n) d[i] = (_Float16)s[i];
}
// strided column-slice conversion: dst[r, 0..w) = src[r, c0..c0+w)
__global__ void cvt_f16_cols(const float* __restrict__ s, _Float16* __restrict__ d,
                             int rows, int srcld, int c0, int w) {
  int i = blockIdx.x * blockDim.x + threadIdx.x;
  if (i >= rows * w) return;
  int r = i / w, c = i % w;
  d[i] = (_Float16)s[(size_t)r * srcld + c0 + c];
}
__global__ void fill_f32(float* p, int n) {
  int i = blockIdx.x * blockDim.x + threadIdx.x;
  if (i < n) p[i] = 0.f;
}
__global__ void fill_f16(_Float16* p, int n) {
  int i = blockIdx.x * blockDim.x + threadIdx.x;
  if (i < n) p[i] = (_Float16)0.f;
}

// GRU gate with optional constant gi part (gib):
// r=sig(gi0+gib0+gh0) z=sig(gi1+gib1+gh1) n=tanh(gi2+gib2+r*gh2) h=(1-z)n+z*hprev
__global__ void gru_gate(const float* __restrict__ gi, const float* __restrict__ gib,
                         const float* __restrict__ gh, const float* __restrict__ hprev,
                         float* __restrict__ hout, _Float16* __restrict__ houth) {
  int idx = blockIdx.x * blockDim.x + threadIdx.x;
  if (idx >= SIZE * HH) return;
  int row = idx >> 8, c = idx & 255;
  size_t b3 = (size_t)row * H3;
  float g0 = gi[b3 + c]          + (gib ? gib[b3 + c]          : 0.f);
  float g1 = gi[b3 + HH + c]     + (gib ? gib[b3 + HH + c]     : 0.f);
  float g2 = gi[b3 + 2 * HH + c] + (gib ? gib[b3 + 2 * HH + c] : 0.f);
  float r = 1.f / (1.f + expf(-(g0 + gh[b3 + c])));
  float z = 1.f / (1.f + expf(-(g1 + gh[b3 + HH + c])));
  float n = tanhf(g2 + r * gh[b3 + 2 * HH + c]);
  float hp = hprev ? hprev[(size_t)row * HH + c] : 0.f;
  float h = (1.f - z) * n + z * hp;
  hout[(size_t)row * HH + c] = h;
  houth[(size_t)row * HH + c] = (_Float16)h;
}

// gathered half of seq[t]: seqg[b*8+i] = h[b, colidx(i,t)]  (f16, [8192,256])
__global__ void build_seqg(const float* __restrict__ h, _Float16* __restrict__ seqg, int t) {
  int idx = blockIdx.x * blockDim.x + threadIdx.x;
  if (idx >= SIZE * HH) return;
  int row = idx >> 8, c = idx & 255;
  int b = row >> 3, i = row & 7;
  int j = (t < i) ? t : t + 1;
  seqg[idx] = (_Float16)h[((size_t)(b * 8 + j)) * HH + c];
}

// partial logits: plog[t][row][c] (set on fwd pass, accumulate + bias on bwd)
__global__ void plog_acc(const float* __restrict__ h, const float* __restrict__ Wh,
                         const float* __restrict__ bh, float* __restrict__ plog,
                         int t, int colofs, int mode) {
  int row = blockIdx.x * blockDim.x + threadIdx.x;
  if (row >= SIZE) return;
  const float* hr = h + (size_t)row * HH;
  float d0 = 0.f, d1 = 0.f;
  for (int c = 0; c < HH; ++c) {
    float hv = hr[c];
    d0 += hv * Wh[colofs + c];
    d1 += hv * Wh[H2 + colofs + c];
  }
  size_t o = (size_t)t * SIZE * 2 + (size_t)row * 2;
  if (mode == 0) { plog[o] = d0; plog[o + 1] = d1; }
  else          { plog[o] += d0 + bh[0]; plog[o + 1] += d1 + bh[1]; }
}

// ---------------------------------------------------------------------------
// JAX threefry2x32 (key(42) -> [0,42]) -> gumbel -> hard weights (tau softmax)
// ---------------------------------------------------------------------------
__device__ __forceinline__ unsigned rotl32(unsigned x, int d) { return (x << d) | (x >> (32 - d)); }
__device__ __forceinline__ void threefry(unsigned x0, unsigned x1, unsigned& o0, unsigned& o1) {
  const unsigned ks0 = 0u, ks1 = 42u, ks2 = ks0 ^ ks1 ^ 0x1BD11BDAu;
  x0 += ks0; x1 += ks1;
#define TF4(a,b,c,d) x0+=x1; x1=rotl32(x1,a); x1^=x0; x0+=x1; x1=rotl32(x1,b); x1^=x0; \
                     x0+=x1; x1=rotl32(x1,c); x1^=x0; x0+=x1; x1=rotl32(x1,d); x1^=x0;
  TF4(13,15,26,6)  x0 += ks1; x1 += ks2 + 1u;
  TF4(17,29,16,24) x0 += ks2; x1 += ks0 + 2u;
  TF4(13,15,26,6)  x0 += ks0; x1 += ks1 + 3u;
  TF4(17,29,16,24) x0 += ks1; x1 += ks2 + 4u;
  TF4(13,15,26,6)  x0 += ks2; x1 += ks0 + 5u;
#undef TF4
  o0 = x0; o1 = x1;
}
__device__ __forceinline__ float gumbel_at(unsigned f) {
  const unsigned half = (unsigned)(SIZE * LL);          // 57344
  unsigned j = (f < half) ? f : f - half;
  unsigned o0, o1; threefry(j, j + half, o0, o1);
  unsigned bits = (f < half) ? o0 : o1;
  float u = __uint_as_float((bits >> 9) | 0x3F800000u) - 1.0f;
  const float tiny = 1.17549435e-38f;
  u = fmaxf(tiny, u * (1.0f - tiny) + tiny);
  return -logf(-logf(u));
}
__global__ void gumbel_hw(const float* __restrict__ plog, float* __restrict__ hwm) {
  int idx = blockIdx.x * blockDim.x + threadIdx.x;  // == r*7 + t
  if (idx >= SIZE * LL) return;
  int r = idx / LL, t = idx % LL;
  size_t o = (size_t)t * SIZE * 2 + (size_t)r * 2;
  float l0 = plog[o], l1 = plog[o + 1];
  float g0 = gumbel_at((unsigned)(2 * idx));
  float g1 = gumbel_at((unsigned)(2 * idx + 1));
  float a = ((l1 + g1) - (l0 + g0)) / TAU_F;
  float y1 = 1.f / (1.f + expf(-a));
  int b = r >> 3, i = r & 7;
  int j = (t < i) ? t : t + 1;
  hwm[(size_t)b * 64 + i * 8 + j] = y1;
}

// ---------------------------------------------------------------------------
// 8-agent attention: one wave32 per (b,i); fused with f16 concat build
// ---------------------------------------------------------------------------
__global__ __launch_bounds__(32)
void attn_kernel(const float* __restrict__ q, const float* __restrict__ k,
                 const float* __restrict__ v, const float* __restrict__ hwm,
                 const float* __restrict__ hout, _Float16* __restrict__ cat) {
  int row = blockIdx.x, lane = threadIdx.x;
  int b = row >> 3, i = row & 7;
  const float* qr = q + (size_t)row * AA;
  float qv[4];
#pragma unroll
  for (int u = 0; u < 4; ++u) qv[u] = qr[lane * 4 + u];
  float s[NAG];
#pragma unroll
  for (int j = 0; j < NAG; ++j) {
    const float* kr = k + ((size_t)(b * 8 + j)) * AA;
    float p = 0.f;
#pragma unroll
    for (int u = 0; u < 4; ++u) p += qv[u] * kr[lane * 4 + u];
    for (int off = 16; off >= 1; off >>= 1) p += __shfl_xor(p, off, 32);
    s[j] = (j == i) ? -1e9f : p * 0.08838834764831845f;   // 1/sqrt(128)
  }
  float mx = s[0];
#pragma unroll
  for (int j = 1; j < NAG; ++j) mx = fmaxf(mx, s[j]);
  float e[NAG], den = 0.f;
#pragma unroll
  for (int j = 0; j < NAG; ++j) { e[j] = expf(s[j] - mx); den += e[j]; }
  float x[4] = {0.f, 0.f, 0.f, 0.f};
#pragma unroll
  for (int j = 0; j < NAG; ++j) {
    float w = (e[j] / den) * hwm[(size_t)b * 64 + i * 8 + j];
    const float* vr = v + ((size_t)(b * 8 + j)) * AA;
#pragma unroll
    for (int u = 0; u < 4; ++u) x[u] += w * vr[lane * 4 + u];
  }
  _Float16* crow = cat + (size_t)row * (HH + AA);
#pragma unroll
  for (int u = 0; u < 4; ++u) crow[HH + lane * 4 + u] = (_Float16)x[u];
  const float* hr = hout + (size_t)row * HH;
#pragma unroll
  for (int u = 0; u < 8; ++u) crow[lane * 8 + u] = (_Float16)hr[lane * 8 + u];
}

// ---------------------------------------------------------------------------
// Host orchestration
// ---------------------------------------------------------------------------
static inline void* bump(char*& p, size_t bytes) {
  void* r = p; p += (bytes + 255) & ~(size_t)255; return r;
}
static void gemm(hipStream_t s, const _Float16* A, const _Float16* B, const float* bias,
                 float* Cf, _Float16* Ch, int M, int N, int K, int ldc, int ldch, int act) {
  if (N % 64 == 0) {
    dim3 g(M / 128, N / 64);
    wmma_gemm<4><<<g, 128, 0, s>>>(A, B, bias, Cf, Ch, K, ldc, ldch, act);
  } else {
    dim3 g(M / 128, N / 32);
    wmma_gemm<2><<<g, 128, 0, s>>>(A, B, bias, Cf, Ch, K, ldc, ldch, act);
  }
}
static void cvt(hipStream_t s, const float* src, _Float16* dst, int n) {
  cvt_f16<<<(n + 255) / 256, 256, 0, s>>>(src, dst, n);
}

extern "C" void kernel_launch(void* const* d_in, const int* in_sizes, int n_in,
                              void* d_out, int out_size, void* d_ws, size_t ws_size,
                              hipStream_t stream) {
  const float* obs   = (const float*)d_in[0];
  const float* hid   = (const float*)d_in[1];
  const float* Wenc  = (const float*)d_in[2];
  const float* benc  = (const float*)d_in[3];
  const float* Wihc  = (const float*)d_in[4];
  const float* Whhc  = (const float*)d_in[5];
  const float* bihc  = (const float*)d_in[6];
  const float* bhhc  = (const float*)d_in[7];
  const float* Wihf  = (const float*)d_in[8];
  const float* Whhf  = (const float*)d_in[9];
  const float* bihf  = (const float*)d_in[10];
  const float* bhhf  = (const float*)d_in[11];
  const float* Wihb  = (const float*)d_in[12];
  const float* Whhb  = (const float*)d_in[13];
  const float* bihb  = (const float*)d_in[14];
  const float* bhhb  = (const float*)d_in[15];
  const float* Whard = (const float*)d_in[16];
  const float* bhard = (const float*)d_in[17];
  const float* Wq    = (const float*)d_in[18];
  const float* Wk    = (const float*)d_in[19];
  const float* Wv    = (const float*)d_in[20];
  const float* bv    = (const float*)d_in[21];
  const float* Wdec  = (const float*)d_in[22];
  const float* bdec  = (const float*)d_in[23];

  float* out_logits = (float*)d_out;                  // [8192, 32]
  float* h_out      = (float*)d_out + SIZE * NACT;    // [8192, 256] (2nd output)

  char* p = (char*)d_ws;
  _Float16* wenc_h  = (_Float16*)bump(p, (size_t)HH * INDIM * 2);
  _Float16* wihc_h  = (_Float16*)bump(p, (size_t)H3 * HH * 2);
  _Float16* whhc_h  = (_Float16*)bump(p, (size_t)H3 * HH * 2);
  _Float16* wihf0_h = (_Float16*)bump(p, (size_t)H3 * HH * 2);  // Wihf[:, 0:256]
  _Float16* wihf1_h = (_Float16*)bump(p, (size_t)H3 * HH * 2);  // Wihf[:, 256:512]
  _Float16* whhf_h  = (_Float16*)bump(p, (size_t)H3 * HH * 2);
  _Float16* wihb0_h = (_Float16*)bump(p, (size_t)H3 * HH * 2);
  _Float16* wihb1_h = (_Float16*)bump(p, (size_t)H3 * HH * 2);
  _Float16* whhb_h  = (_Float16*)bump(p, (size_t)H3 * HH * 2);
  _Float16* wq_h    = (_Float16*)bump(p, (size_t)AA * HH * 2);
  _Float16* wk_h    = (_Float16*)bump(p, (size_t)AA * HH * 2);
  _Float16* wv_h    = (_Float16*)bump(p, (size_t)AA * HH * 2);
  _Float16* wdec_h  = (_Float16*)bump(p, (size_t)NACT * (HH + AA) * 2);
  _Float16* obs_h   = (_Float16*)bump(p, (size_t)SIZE * INDIM * 2);
  _Float16* hid_h   = (_Float16*)bump(p, (size_t)SIZE * HH * 2);
  _Float16* enc_h   = (_Float16*)bump(p, (size_t)SIZE * HH * 2);
  _Float16* hout_h  = (_Float16*)bump(p, (size_t)SIZE * HH * 2);
  _Float16* hcur_h  = (_Float16*)bump(p, (size_t)SIZE * HH * 2);
  _Float16* seqg_h  = (_Float16*)bump(p, (size_t)SIZE * HH * 2);
  _Float16* cat_h   = (_Float16*)bump(p, (size_t)SIZE * (HH + AA) * 2);
  float* gi   = (float*)bump(p, (size_t)SIZE * H3 * 4);
  float* gib  = (float*)bump(p, (size_t)SIZE * H3 * 4);   // t-invariant gi part
  float* gh   = (float*)bump(p, (size_t)SIZE * H3 * 4);
  float* hcur = (float*)bump(p, (size_t)SIZE * HH * 4);
  float* plog = (float*)bump(p, (size_t)LL * SIZE * 2 * 4);
  float* hwm  = (float*)bump(p, (size_t)BSZ * NAG * NAG * 4);
  float* qb   = (float*)bump(p, (size_t)SIZE * AA * 4);
  float* kb   = (float*)bump(p, (size_t)SIZE * AA * 4);
  float* vb   = (float*)bump(p, (size_t)SIZE * AA * 4);

  // --- weight / input conversion to f16 (weights stay hot in 192MB L2) ---
  cvt(stream, Wenc, wenc_h, HH * INDIM);
  cvt(stream, Wihc, wihc_h, H3 * HH);
  cvt(stream, Whhc, whhc_h, H3 * HH);
  cvt_f16_cols<<<(H3 * HH + 255) / 256, 256, 0, stream>>>(Wihf, wihf0_h, H3, H2, 0,  HH);
  cvt_f16_cols<<<(H3 * HH + 255) / 256, 256, 0, stream>>>(Wihf, wihf1_h, H3, H2, HH, HH);
  cvt(stream, Whhf, whhf_h, H3 * HH);
  cvt_f16_cols<<<(H3 * HH + 255) / 256, 256, 0, stream>>>(Wihb, wihb0_h, H3, H2, 0,  HH);
  cvt_f16_cols<<<(H3 * HH + 255) / 256, 256, 0, stream>>>(Wihb, wihb1_h, H3, H2, HH, HH);
  cvt(stream, Whhb, whhb_h, H3 * HH);
  cvt(stream, Wq,   wq_h,   AA * HH);
  cvt(stream, Wk,   wk_h,   AA * HH);
  cvt(stream, Wv,   wv_h,   AA * HH);
  cvt(stream, Wdec, wdec_h, NACT * (HH + AA));
  cvt(stream, obs,  obs_h,  SIZE * INDIM);
  cvt(stream, hid,  hid_h,  SIZE * HH);

  // --- encoder: enc = relu(obs @ Wenc^T + benc), f16 output ---
  gemm(stream, obs_h, wenc_h, benc, nullptr, enc_h, SIZE, HH, INDIM, HH, HH, 1);

  // --- comm GRU cell: h_out = gru(enc, hidden) ---
  gemm(stream, enc_h, wihc_h, bihc, gi, nullptr, SIZE, H3, HH, H3, H3, 0);
  gemm(stream, hid_h, whhc_h, bhhc, gh, nullptr, SIZE, H3, HH, H3, H3, 0);
  gru_gate<<<SIZE * HH / 256, 256, 0, stream>>>(gi, nullptr, gh, hid, h_out, hout_h);

  // --- forward GRU scan (7 steps); gib = h_i part, computed once ---
  gemm(stream, hout_h, wihf0_h, bihf, gib, nullptr, SIZE, H3, HH, H3, H3, 0);
  fill_f32<<<SIZE * HH / 256, 256, 0, stream>>>(hcur, SIZE * HH);
  fill_f16<<<SIZE * HH / 256, 256, 0, stream>>>(hcur_h, SIZE * HH);
  for (int t = 0; t < LL; ++t) {
    build_seqg<<<SIZE * HH / 256, 256, 0, stream>>>(h_out, seqg_h, t);
    gemm(stream, seqg_h, wihf1_h, nullptr, gi, nullptr, SIZE, H3, HH, H3, H3, 0);
    gemm(stream, hcur_h, whhf_h, bhhf, gh, nullptr, SIZE, H3, HH, H3, H3, 0);
    gru_gate<<<SIZE * HH / 256, 256, 0, stream>>>(gi, gib, gh, hcur, hcur, hcur_h);
    plog_acc<<<SIZE / 256, 256, 0, stream>>>(hcur, Whard, bhard, plog, t, 0, 0);
  }
  // --- backward GRU scan (process seq[6]..seq[0]; hb[t] = state after seq[t]) ---
  gemm(stream, hout_h, wihb0_h, bihb, gib, nullptr, SIZE, H3, HH, H3, H3, 0);
  fill_f32<<<SIZE * HH / 256, 256, 0, stream>>>(hcur, SIZE * HH);
  fill_f16<<<SIZE * HH / 256, 256, 0, stream>>>(hcur_h, SIZE * HH);
  for (int s = 0; s < LL; ++s) {
    int t = LL - 1 - s;
    build_seqg<<<SIZE * HH / 256, 256, 0, stream>>>(h_out, seqg_h, t);
    gemm(stream, seqg_h, wihb1_h, nullptr, gi, nullptr, SIZE, H3, HH, H3, H3, 0);
    gemm(stream, hcur_h, whhb_h, bhhb, gh, nullptr, SIZE, H3, HH, H3, H3, 0);
    gru_gate<<<SIZE * HH / 256, 256, 0, stream>>>(gi, gib, gh, hcur, hcur, hcur_h);
    plog_acc<<<SIZE / 256, 256, 0, stream>>>(hcur, Whard, bhard, plog, t, HH, 1);
  }

  // --- Gumbel hard attention weights (exact JAX threefry, key(42)) ---
  fill_f32<<<(BSZ * 64 + 255) / 256, 256, 0, stream>>>(hwm, BSZ * 64);
  gumbel_hw<<<(SIZE * LL + 255) / 256, 256, 0, stream>>>(plog, hwm);

  // --- q/k/v projections ---
  gemm(stream, hout_h, wq_h, nullptr, qb, nullptr, SIZE, AA, HH, AA, AA, 0);
  gemm(stream, hout_h, wk_h, nullptr, kb, nullptr, SIZE, AA, HH, AA, AA, 0);
  gemm(stream, hout_h, wv_h, bv,      vb, nullptr, SIZE, AA, HH, AA, AA, 1);

  // --- soft*hard attention + build concat(h_out, x) in f16 ---
  attn_kernel<<<SIZE, 32, 0, stream>>>(qb, kb, vb, hwm, h_out, cat_h);

  // --- decoder: output = cat @ Wdec^T + bdec ---
  gemm(stream, cat_h, wdec_h, bdec, out_logits, nullptr, SIZE, NACT, HH + AA, NACT, NACT, 0);
}
